// RelRepContextBase_53145925320931
// MI455X (gfx1250) — matrix operand order repaired and checked by reading, not compiled, not run on hardware
//
#include <hip/hip_runtime.h>
#include <hip/hip_bf16.h>
#include <stdint.h>

// ---------------------------------------------------------------------------
// Problem constants (from reference): B=2, S=2048, H=256, NS=512, NR=512
// d_in = 3H = 768, d_ff = 4*d_in = 3072, rows M = B*NR = 1024
// ---------------------------------------------------------------------------
#define BB   2
#define SS   2048
#define HH   256
#define NSP  512
#define NRL  512
#define MR   (BB * NRL)     // 1024 relation rows
#define DIN  (3 * HH)       // 768
#define DFF  (4 * DIN)      // 3072

typedef __bf16 bf16t;
typedef __attribute__((ext_vector_type(8)))  __bf16 v8bf;
typedef __attribute__((ext_vector_type(16))) __bf16 v16bf;
typedef __attribute__((ext_vector_type(8)))  float  v8f;

// ---------------------------------------------------------------------------
// CDNA5 async global->LDS copy (ASYNCcnt-tracked, ISA 08_async_tensor §4).
// VDST = VGPR holding LDS byte address, VADDR = 64-bit global address.
// Generic shared pointers carry the raw LDS offset in their low 32 bits.
// ---------------------------------------------------------------------------
__device__ __forceinline__ void async_copy_b128(uint32_t lds_byte_addr,
                                                const void* gptr) {
    asm volatile("global_load_async_to_lds_b128 %0, %1, off"
                 :: "v"(lds_byte_addr), "v"((uint64_t)(uintptr_t)gptr)
                 : "memory");
}
__device__ __forceinline__ void wait_async_le1() {
    asm volatile("s_wait_asynccnt 0x1" ::: "memory");
}
__device__ __forceinline__ void wait_async_le0() {
    asm volatile("s_wait_asynccnt 0x0" ::: "memory");
}

// ---------------------------------------------------------------------------
// Kernel 1: gather head/tail span reps + masked max-pool of context window,
// emit bf16 feature rows X[r, 0:768] = [head | tail | ctx].
// One block per relation; thread == channel. Window bounds are wave-uniform.
// ---------------------------------------------------------------------------
__global__ void build_x_kernel(const float*   __restrict__ token_reps,   // (B,S,H)
                               const uint8_t* __restrict__ token_masks,  // (B,S) bool
                               const float*   __restrict__ span_reps,    // (B,NS,H)
                               const int*     __restrict__ span_ids,     // (B,NS,2)
                               const int*     __restrict__ rel_ids,      // (B,NR,2)
                               const uint8_t* __restrict__ rel_masks,    // (B,NR) bool
                               const float*   __restrict__ neg_limit_p,  // scalar
                               bf16t*         __restrict__ Xbf)          // (M, DIN)
{
    const int r  = blockIdx.x;          // 0 .. B*NR-1
    const int b  = r / NRL;
    const int rr = r - b * NRL;
    const int c  = threadIdx.x;         // channel 0..255
    const float neg_limit = neg_limit_p[0];

    const int head = rel_ids[(b * NRL + rr) * 2 + 0];
    const int tail = rel_ids[(b * NRL + rr) * 2 + 1];
    const int hs = span_ids[(b * NSP + head) * 2 + 0];
    const int he = span_ids[(b * NSP + head) * 2 + 1];
    const int ts = span_ids[(b * NSP + tail) * 2 + 0];
    const int te = span_ids[(b * NSP + tail) * 2 + 1];
    const bool relm = rel_masks[b * NRL + rr] != 0;

    const int lo = min(he, te);
    const int hi = max(hs, ts);

    float acc = neg_limit;
    bool  any = false;
    if (relm) {
        for (int s = lo; s < hi; ++s) {
            if (s + 8 < hi)   // global_prefetch_b8 a future token row element
                __builtin_prefetch(&token_reps[((size_t)b * SS + s + 8) * HH + c], 0, 0);
            const bool in_head = (s >= hs) && (s < he);
            const bool in_tail = (s >= ts) && (s < te);
            if (!in_head && !in_tail && token_masks[b * SS + s]) {
                const float v = token_reps[((size_t)b * SS + s) * HH + c];
                acc = fmaxf(acc, v);
                any = true;
            }
        }
    }
    const float ctx = any ? acc : 0.0f;  // all-masked fallback -> zero vector

    const float hrep = span_reps[((size_t)b * NSP + head) * HH + c];
    const float trep = span_reps[((size_t)b * NSP + tail) * HH + c];

    bf16t* xrow = Xbf + (size_t)r * DIN;
    xrow[c]          = (bf16t)hrep;
    xrow[HH + c]     = (bf16t)trep;
    xrow[2 * HH + c] = (bf16t)ctx;
}

// ---------------------------------------------------------------------------
// Kernel 2: fp32 (K,N) weight -> bf16 (N,K) transposed copy, so the WMMA B
// operand reads 32 contiguous bytes of K per lane.
// ---------------------------------------------------------------------------
__global__ void convert_wT_kernel(const float* __restrict__ W,  // (K,N)
                                  bf16t*       __restrict__ WT, // (N,K)
                                  int K, int N)
{
    const int idx = blockIdx.x * blockDim.x + threadIdx.x;
    if (idx < K * N) {
        const int k = idx / N;
        const int n = idx - k * N;
        WT[(size_t)n * K + k] = (bf16t)W[idx];
    }
}

// ---------------------------------------------------------------------------
// Kernel 3: bf16 WMMA GEMM  C = act(A(MxK) * BT(NxK)^T + bias)
//
// Block = 8 waves -> 64(M) x 128(N) tile. K stepped by 32.
// Shared 64x32 A tile staged in LDS via double-buffered async global->LDS
// copies (1 instr/wave/K-step -> exact ASYNCcnt bookkeeping; async loads
// complete in order, so after issuing the next batch, wait<=1 retires the
// previous one). Last K-step peeled so the steady-state loop body is
// branch-free (unconditional issue + s_wait_asynccnt 0x1).
// A sched_barrier(0) between the fragment loads and the WMMA chain forces
// all 8 ds_load_b128 to issue before any WMMA: one s_wait_dscnt per K-step
// and 4 back-to-back WMMAs instead of per-tile load->wait->wmma.
// LDS rows padded to 80B (20 banks) -> conflict-free 16-lane fragment reads.
//
// Fragment layouts used here (wave32; A/B in contiguous K order, C/D as in ISA 7.12.2):
//   A 16x32: lanes 0-15 row M=l, K = 16*half..16*half+15
//   B 32x16: lanes 0-15 col N=l hold K=0..15; lanes 16-31 hold K=16..31
//   C/D:     VGPR i -> row m0+i+8*half, col n0+(lane&15)
// EXEC is all ones everywhere (only uniform scalar control flow).
// ---------------------------------------------------------------------------
#define LDS_ROW_B 80                       // 64B of data + 16B pad
#define LDS_BUF_B (64 * LDS_ROW_B)         // 5120B per buffer

template <bool RELU, bool OUT_BF16>
__global__ void gemm_wmma_bf16(const bf16t* __restrict__ A,    // (M,K) row-major
                               const bf16t* __restrict__ BT,   // (N,K) row-major
                               const float* __restrict__ bias, // (N)
                               void*        __restrict__ Cout, // (M,N)
                               int M, int N, int K)
{
    __shared__ uint8_t smem[2 * LDS_BUF_B];

    constexpr int MT = 4;                       // 16-row tiles per wave
    const int tid     = threadIdx.x;
    const int lane    = tid & 31;
    const int wave    = tid >> 5;
    const int n0      = blockIdx.x * 128 + wave * 16;
    const int m0      = blockIdx.y * (16 * MT);
    const int halfsel = lane >> 4;              // 0: lanes 0-15, 1: lanes 16-31
    const int l16     = lane & 15;
    const int abase   = 16 * halfsel;           // A K-chunk base within 32
    const int bbase   = 16 * halfsel;           // B K-chunk base within 32

    // cooperative A staging: thread t copies 16B: row t/4, 8-elem chunk t%4
    const int arow   = tid >> 2;                // 0..63
    const int achunk = tid & 3;                 // 0..3
    const uint32_t lds_base   = (uint32_t)(uintptr_t)&smem[0];
    const uint32_t my_lds_off = (uint32_t)(arow * LDS_ROW_B + achunk * 16);
    const bf16t*   agsrc      = A + (size_t)(m0 + arow) * K + achunk * 8;

    v8f acc[MT];
#pragma unroll
    for (int t = 0; t < MT; ++t) acc[t] = (v8f)(0.0f);

    const bf16t* bcolp = BT + (size_t)(n0 + l16) * K + bbase;

    // compute step: 8 ds_load_b128 first, then 4 back-to-back WMMAs.
    auto compute_step = [&](int buf, const v16bf& bv) {
        const uint8_t* abuf = smem + buf * LDS_BUF_B;
        union { v16bf v; v8bf h[2]; } afrag[MT];
#pragma unroll
        for (int t = 0; t < MT; ++t) {
            const uint8_t* ap = abuf + (t * 16 + l16) * LDS_ROW_B + abase * 2;
            afrag[t].h[0] = *(const v8bf*)(ap);        // K = base .. base+7
            afrag[t].h[1] = *(const v8bf*)(ap + 16);   // K = base+8 .. base+15
        }
        // keep all ds_loads before all WMMAs (single s_wait_dscnt exposure)
        __builtin_amdgcn_sched_barrier(0);
#pragma unroll
        for (int t = 0; t < MT; ++t) {
            acc[t] = __builtin_amdgcn_wmma_f32_16x16x32_bf16(
                false, afrag[t].v, false, bv, (short)0, acc[t], false, false);
        }
    };

    // prologue: fill buffer 0 for k = 0
    async_copy_b128(lds_base + my_lds_off, agsrc);

    int buf = 0;
    int k = 0;
    // steady state: branch-free issue + wait<=1
    for (; k + 32 < K; k += 32) {
        async_copy_b128(lds_base + (uint32_t)((buf ^ 1) * LDS_BUF_B) + my_lds_off,
                        agsrc + (k + 32));

        union { v16bf v; v8bf h[2]; } bfrag;    // per-wave-unique B fragment
        bfrag.v = *(const v16bf*)(bcolp + k);   // 32B contiguous per lane

        wait_async_le1();                        // previous batch retired
        __syncthreads();                         // A tile visible to all waves

        compute_step(buf, bfrag.v);

        __syncthreads();                         // done reading before overwrite
        buf ^= 1;
    }
    // epilogue: last K-step (no further issue)
    {
        union { v16bf v; v8bf h[2]; } bfrag;
        bfrag.v = *(const v16bf*)(bcolp + k);
        wait_async_le0();
        __syncthreads();
        compute_step(buf, bfrag.v);
    }

    const int   col = n0 + l16;
    const float bv  = bias[col];
#pragma unroll
    for (int t = 0; t < MT; ++t) {
#pragma unroll
        for (int i = 0; i < 8; ++i) {
            const int row = m0 + t * 16 + i + 8 * halfsel;
            float v = acc[t][i] + bv;
            if (RELU) v = fmaxf(v, 0.0f);
            if (OUT_BF16) ((bf16t*)Cout)[(size_t)row * N + col] = (bf16t)v;
            else          ((float*)Cout)[(size_t)row * N + col] = v;
        }
    }
}

// ---------------------------------------------------------------------------
// Launch: X build -> weight converts -> GEMM1(relu, bf16 out) -> GEMM2(f32 out)
// Workspace: Xbf 1.5MB + W1T 4.5MB + W2T 1.5MB + Hbf 6MB  (~13.5MB, 256B aligned)
// ---------------------------------------------------------------------------
extern "C" void kernel_launch(void* const* d_in, const int* in_sizes, int n_in,
                              void* d_out, int out_size, void* d_ws, size_t ws_size,
                              hipStream_t stream)
{
    const float*   token_reps  = (const float*)d_in[0];
    const uint8_t* token_masks = (const uint8_t*)d_in[1];   // jnp bool -> 1 byte
    const float*   span_reps   = (const float*)d_in[2];
    const int*     span_ids    = (const int*)d_in[3];
    const int*     rel_ids     = (const int*)d_in[4];
    const uint8_t* rel_masks   = (const uint8_t*)d_in[5];   // jnp bool -> 1 byte
    const float*   neg_limit   = (const float*)d_in[6];
    const float*   W1          = (const float*)d_in[7];
    const float*   b1          = (const float*)d_in[8];
    const float*   W2          = (const float*)d_in[9];
    const float*   b2          = (const float*)d_in[10];
    float*         out         = (float*)d_out;             // (B, NR, H) fp32

    uint8_t* ws = (uint8_t*)d_ws;
    size_t off = 0;
    auto alloc = [&](size_t bytes) -> uint8_t* {
        uint8_t* p = ws + off;
        off += (bytes + 255) & ~(size_t)255;
        return p;
    };
    bf16t* Xbf = (bf16t*)alloc((size_t)MR  * DIN * sizeof(bf16t));
    bf16t* W1T = (bf16t*)alloc((size_t)DFF * DIN * sizeof(bf16t));
    bf16t* W2T = (bf16t*)alloc((size_t)HH  * DFF * sizeof(bf16t));
    bf16t* Hbf = (bf16t*)alloc((size_t)MR  * DFF * sizeof(bf16t));

    // 1) feature rows (gather + masked max-pool) -> bf16 X
    build_x_kernel<<<MR, HH, 0, stream>>>(token_reps, token_masks, span_reps,
                                          span_ids, rel_ids, rel_masks,
                                          neg_limit, Xbf);

    // 2) weights -> transposed bf16
    convert_wT_kernel<<<(DIN * DFF + 255) / 256, 256, 0, stream>>>(W1, W1T, DIN, DFF);
    convert_wT_kernel<<<(DFF * HH  + 255) / 256, 256, 0, stream>>>(W2, W2T, DFF, HH);

    // 3) H = relu(X @ W1 + b1)  : M=1024, N=3072, K=768
    {
        dim3 grid(DFF / 128, MR / 64);
        gemm_wmma_bf16<true, true><<<grid, 256, 0, stream>>>(Xbf, W1T, b1, Hbf,
                                                             MR, DFF, DIN);
    }
    // 4) out = H @ W2 + b2      : M=1024, N=256, K=3072
    {
        dim3 grid(HH / 128, MR / 64);
        gemm_wmma_bf16<false, false><<<grid, 256, 0, stream>>>(Hbf, W2T, b2, out,
                                                               MR, HH, DFF);
    }
}
